// FusedMoEIntegrator_25580825215317
// MI455X (gfx1250) — compile-verified, matrix operand order checked by reading, not run on hardware
//
#include <hip/hip_runtime.h>

// ---------------------------------------------------------------------------
// FusedMoEIntegrator for gfx1250 (MI455X).
// Compute-bound (~0.5 TFLOP vs ~150MB traffic) -> all GEMMs via
// v_wmma_f32_16x16x32_bf16 (bf16 inputs, f32 accumulate).
// Activations staged as bf16 in LDS once; A-frags are pure uint4 DS loads.
// x_next carried in bf16 (it is only ever a GEMM input downstream).
// ---------------------------------------------------------------------------

#define D_MODEL 1024
#define CTX_DIM 2048
#define HID 64
#define NEXP 8
#define DT_CONST 0.1f

typedef __attribute__((ext_vector_type(16))) __bf16 v16bf;
typedef __attribute__((ext_vector_type(8)))  float  v8f;

union FragAB {
  v16bf v;
  uint4 q[2];
  unsigned short u[16];
};

__device__ __forceinline__ unsigned short f2bf(float f) {
  unsigned u = __float_as_uint(f);
  u += 0x7FFFu + ((u >> 16) & 1u);      // round-to-nearest-even
  return (unsigned short)(u >> 16);
}
__device__ __forceinline__ void store_bf16x4(unsigned short* dst, float4 f) {
  uint2 u;
  u.x = (unsigned)f2bf(f.x) | ((unsigned)f2bf(f.y) << 16);
  u.y = (unsigned)f2bf(f.z) | ((unsigned)f2bf(f.w) << 16);
  *reinterpret_cast<uint2*>(dst) = u;
}
__device__ __forceinline__ float geluf(float x) {
  return 0.5f * x * (1.0f + erff(x * 0.70710678118654752f));
}
__device__ __forceinline__ float sigm(float x) { return 1.0f / (1.0f + expf(-x)); }
__device__ __forceinline__ float softplusf(float x) { return (x > 20.f) ? x : log1pf(expf(x)); }

__device__ __forceinline__ v8f vzero() {
  v8f z;
#pragma unroll
  for (int i = 0; i < 8; ++i) z[i] = 0.f;
  return z;
}

__device__ __forceinline__ v8f wmma_bf16(const FragAB& a, const FragAB& b, v8f c) {
  return __builtin_amdgcn_wmma_f32_16x16x32_bf16(false, a.v, false, b.v, (short)0, c,
                                                 false, false);
}

// A or B fragment from bf16 (ushort) storage laid out [rows][stride]; row = n.
// Element e of lane l maps to k = k0 + (e&8?16:0) + (l>>4)*8 + (e&7).
__device__ __forceinline__ void load_bf16(FragAB& f, const unsigned short* base,
                                          int stride, int n, int k0, int hf) {
  const unsigned short* p = base + (size_t)n * stride + k0 + hf * 8;
  f.q[0] = *reinterpret_cast<const uint4*>(p);
  f.q[1] = *reinterpret_cast<const uint4*>(p + 16);
}

// One shared A row tile (16xK from LDS) against NB consecutive 16-col B tiles.
// nlane already includes (lane&15).
template <int NB>
__device__ __forceinline__ void gemm_rowtiles(v8f acc[NB],
                                              const unsigned short* A, int astride, int m,
                                              const unsigned short* B, int bstride, int nlane,
                                              int K, int hf) {
  for (int k0 = 0; k0 < K; k0 += 32) {
    FragAB a;
    load_bf16(a, A, astride, m, k0, hf);
#pragma unroll
    for (int j = 0; j < NB; ++j) {
      FragAB b;
      load_bf16(b, B, bstride, nlane + j * 16, k0, hf);
      acc[j] = wmma_bf16(a, b, acc[j]);
    }
  }
}

// ---------------------------------------------------------------------------
// Weight convert + transpose: in[b][r][c] (f32) -> out[b][c][r] (bf16)
// ---------------------------------------------------------------------------
__global__ void k_transpose_bf16(const float* __restrict__ in,
                                 unsigned short* __restrict__ out,
                                 int B, int R, int C) {
  long long total = (long long)B * R * C;
  long long idx = (long long)blockIdx.x * blockDim.x + threadIdx.x;
  if (idx >= total) return;
  int c = (int)(idx % C);
  long long t = idx / C;
  int r = (int)(t % R);
  int b = (int)(t / R);
  out[((long long)b * C + c) * R + r] = f2bf(in[idx]);
}

// ---------------------------------------------------------------------------
// Router + init: integrated = x, v = 0, dense renormalized top-2 weights.
// ---------------------------------------------------------------------------
__global__ void __launch_bounds__(256) k_router_init(
    const float* __restrict__ x,
    const unsigned short* __restrict__ wr1t,   // [256][1024] bf16
    const float* __restrict__ br1,
    const float* __restrict__ wr2,             // [256][8] f32
    const float* __restrict__ br2,
    float* __restrict__ integ,
    float* __restrict__ v,
    float* __restrict__ wexp) {                // [T][8]
  __shared__ __align__(16) unsigned short sxb[16][1024];   // 32 KB
  __shared__ float sh[16][256];                            // 16 KB
  __shared__ float slog[16][8];
  const int t0 = blockIdx.x * 16;
  const int tid = threadIdx.x;
  const int lane = tid & 31, wid = tid >> 5;
  for (int i = tid; i < 16 * 256; i += 256) {       // float4-granular staging
    int mm = i >> 8, d4 = (i & 255) * 4;
    long long g = (long long)(t0 + mm) * D_MODEL + d4;
    float4 val = *reinterpret_cast<const float4*>(&x[g]);
    store_bf16x4(&sxb[mm][d4], val);
    *reinterpret_cast<float4*>(&integ[g]) = val;
    float4 z = make_float4(0.f, 0.f, 0.f, 0.f);
    *reinterpret_cast<float4*>(&v[g]) = z;
  }
  __syncthreads();
  const int m = lane & 15, hf = lane >> 4;
  {                                   // 16 tiles: one NB=2 group per wave
    int nlane = wid * 32 + (lane & 15);
    v8f acc[2];
    acc[0] = vzero(); acc[1] = vzero();
    gemm_rowtiles<2>(acc, &sxb[0][0], 1024, m, wr1t, 1024, nlane, 1024, hf);
#pragma unroll
    for (int j = 0; j < 2; ++j) {
      int n = nlane + j * 16;
      float bias = br1[n];
#pragma unroll
      for (int r = 0; r < 8; ++r) sh[r + 8 * hf][n] = geluf(acc[j][r] + bias);
    }
  }
  __syncthreads();
  if (tid < 128) {
    int mm = tid >> 3, e = tid & 7;
    float s = br2[e];
    for (int k = 0; k < 256; ++k) s += sh[mm][k] * wr2[k * 8 + e];
    slog[mm][e] = s;
  }
  __syncthreads();
  if (tid < 16) {
    float mx = slog[tid][0];
    for (int e = 1; e < 8; ++e) mx = fmaxf(mx, slog[tid][e]);
    float p[8], sum = 0.f;
    for (int e = 0; e < 8; ++e) { p[e] = expf(slog[tid][e] - mx); sum += p[e]; }
    for (int e = 0; e < 8; ++e) p[e] /= sum;
    int i1 = 0;
    for (int e = 1; e < 8; ++e) if (p[e] > p[i1]) i1 = e;
    int i2 = (i1 == 0) ? 1 : 0;
    for (int e = 0; e < 8; ++e) if (e != i1 && p[e] > p[i2]) i2 = e;
    float den = p[i1] + p[i2];
    for (int e = 0; e < 8; ++e) {
      float w = (e == i1) ? p[i1] / den : (e == i2) ? p[i2] / den : 0.f;
      wexp[(long long)(t0 + tid) * NEXP + e] = w;
    }
  }
}

// ---------------------------------------------------------------------------
// Expert kernel: shared + all 8 routed experts (dense, weight-masked),
// _inl update and sw-combine fused.  16-token tile, 8 waves.
// ---------------------------------------------------------------------------
__global__ void __launch_bounds__(256) k_experts(
    const float* __restrict__ integ,
    float* __restrict__ v,                     // in/out (in-place)
    unsigned short* __restrict__ xn,           // out, bf16 [T][1024]
    const unsigned short* __restrict__ ws1t,   // [64][2048]
    const float* __restrict__ bs1,
    const unsigned short* __restrict__ ws2t,   // [3072][64]
    const float* __restrict__ bs2,
    const unsigned short* __restrict__ ew1t,   // [8][64][2048]
    const float* __restrict__ eb1,
    const unsigned short* __restrict__ ew2t,   // [8][3072][64]
    const float* __restrict__ eb2,
    const float* __restrict__ wexp,
    const float* __restrict__ shw,
    const float* __restrict__ mu) {
  __shared__ __align__(16) unsigned short sctx[16][2048];        // 64 KB
  __shared__ __align__(16) unsigned short shid[9][16][HID];      // 18 KB
  __shared__ float swt[16][8];
  const int t0 = blockIdx.x * 16;
  const int tid = threadIdx.x;
  const int lane = tid & 31, wid = tid >> 5;
  for (int i = tid; i < 16 * 256; i += 256) {       // float4-granular staging
    int mm = i >> 8, d4 = (i & 255) * 4;
    long long g = (long long)(t0 + mm) * D_MODEL + d4;
    store_bf16x4(&sctx[mm][d4], *reinterpret_cast<const float4*>(&integ[g]));
    store_bf16x4(&sctx[mm][1024 + d4], *reinterpret_cast<const float4*>(&v[g]));
  }
  if (tid < 128) swt[tid >> 3][tid & 7] = wexp[(long long)(t0 + (tid >> 3)) * NEXP + (tid & 7)];
  __syncthreads();
  const float sw = sigm(shw[0]);
  const int m = lane & 15, hf = lane >> 4;

  // Phase 1: hidden[p] = gelu(ctx @ W1_p + b1_p) for p = shared(0), experts 1..8
  // 18 jobs of (path, n-half), NB=2.
  for (int job = wid; job < 18; job += 8) {
    int p = job >> 1, h2 = job & 1;
    const unsigned short* w1 = (p == 0) ? ws1t : (ew1t + (size_t)(p - 1) * HID * CTX_DIM);
    const float* b1 = (p == 0) ? bs1 : (eb1 + (p - 1) * HID);
    int nlane = h2 * 32 + (lane & 15);
    v8f acc[2];
    acc[0] = vzero(); acc[1] = vzero();
    gemm_rowtiles<2>(acc, &sctx[0][0], CTX_DIM, m, w1, CTX_DIM, nlane, CTX_DIM, hf);
#pragma unroll
    for (int j = 0; j < 2; ++j) {
      int n = nlane + j * 16;
      float bias = b1[n];
#pragma unroll
      for (int r = 0; r < 8; ++r) shid[p][r + 8 * hf][n] = f2bf(geluf(acc[j][r] + bias));
    }
  }
  __syncthreads();

  // Phase 2: each wave owns a 128-wide d chunk; per expert: alpha/beta/gate
  // column tiles -> _inl -> weighted accumulate.  ctrl never materialized.
  for (int nt = 0; nt < 8; ++nt) {
    int n = wid * 128 + nt * 16 + (lane & 15);
    float xv[8], vv[8];
#pragma unroll
    for (int r = 0; r < 8; ++r) {
      long long g = (long long)(t0 + r + 8 * hf) * D_MODEL + n;
      xv[r] = integ[g];
      vv[r] = v[g];
    }
    float mun = mu[n];
    v8f ax = vzero(), av = vzero();
    for (int p = 0; p < 9; ++p) {
      const unsigned short* w2 = (p == 0) ? ws2t : (ew2t + (size_t)(p - 1) * 3 * D_MODEL * HID);
      const float* b2 = (p == 0) ? bs2 : (eb2 + (p - 1) * 3 * D_MODEL);
      v8f ca = vzero(), cb = vzero(), cg = vzero();
      for (int k0 = 0; k0 < HID; k0 += 32) {
        FragAB a, bA, bB, bG;
        load_bf16(a, &shid[p][0][0], HID, m, k0, hf);
        load_bf16(bA, w2, HID, n, k0, hf);
        load_bf16(bB, w2, HID, D_MODEL + n, k0, hf);
        load_bf16(bG, w2, HID, 2 * D_MODEL + n, k0, hf);
        ca = wmma_bf16(a, bA, ca);
        cb = wmma_bf16(a, bB, cb);
        cg = wmma_bf16(a, bG, cg);
      }
      float biasA = b2[n], biasB = b2[D_MODEL + n], biasG = b2[2 * D_MODEL + n];
#pragma unroll
      for (int r = 0; r < 8; ++r) {
        int mm = r + 8 * hf;
        float alpha = sigm(ca[r] + biasA);
        float beta  = softplusf(cb[r] + biasB);
        float gate  = sigm(cg[r] + biasG);
        float vnext = alpha * vv[r] - beta * (xv[r] - mun);
        float xnext = xv[r] + DT_CONST * gate * vnext;
        float w = (p == 0) ? sw : (1.0f - sw) * swt[mm][p - 1];
        ax[r] += w * xnext;
        av[r] += w * vnext;
      }
    }
#pragma unroll
    for (int r = 0; r < 8; ++r) {
      long long g = (long long)(t0 + r + 8 * hf) * D_MODEL + n;
      xn[g] = f2bf(ax[r]);
      v[g] = av[r];
    }
  }
}

// ---------------------------------------------------------------------------
// Halt + refine + integrate, fused.  16-token tile, 8 waves.
// ---------------------------------------------------------------------------
__global__ void __launch_bounds__(256) k_refine(
    const unsigned short* __restrict__ xn,      // bf16 [T][1024]
    float* __restrict__ integ,                  // in/out (d_out)
    const unsigned short* __restrict__ wh1t,    // [256][1024]
    const float* __restrict__ bh1,
    const float* __restrict__ wh2,              // [256]
    const float* __restrict__ bh2,              // [1]
    const unsigned short* __restrict__ wf1t,    // [2048][1024]
    const float* __restrict__ bf1v,
    const unsigned short* __restrict__ wf2t,    // [1024][2048]
    const float* __restrict__ bf2v,
    const float* __restrict__ iw) {
  __shared__ __align__(16) unsigned short sxb[16][1024];         // 32 KB
  __shared__ float shh[16][256];                                 // 16 KB
  __shared__ __align__(16) unsigned short shf[16][2048];         // 64 KB
  __shared__ float shalt[16];
  const int t0 = blockIdx.x * 16;
  const int tid = threadIdx.x;
  const int lane = tid & 31, wid = tid >> 5;
  for (int i = tid; i < 16 * 128; i += 256) {       // uint4-granular bf16 copy
    int mm = i >> 7, d8 = (i & 127) * 8;
    *reinterpret_cast<uint4*>(&sxb[mm][d8]) =
        *reinterpret_cast<const uint4*>(&xn[(long long)(t0 + mm) * D_MODEL + d8]);
  }
  __syncthreads();
  const int m = lane & 15, hf = lane >> 4;
  {                                   // halt hidden: 16 tiles, NB=2 per wave
    int nlane = wid * 32 + (lane & 15);
    v8f acc[2];
    acc[0] = vzero(); acc[1] = vzero();
    gemm_rowtiles<2>(acc, &sxb[0][0], 1024, m, wh1t, 1024, nlane, 1024, hf);
#pragma unroll
    for (int j = 0; j < 2; ++j) {
      int n = nlane + j * 16;
      float bias = bh1[n];
#pragma unroll
      for (int r = 0; r < 8; ++r) shh[r + 8 * hf][n] = geluf(acc[j][r] + bias);
    }
  }
  // refine hidden: 128 tiles -> 32 groups of NB=4
  for (int g = wid; g < 32; g += 8) {
    int nlane = g * 64 + (lane & 15);
    v8f acc[4];
#pragma unroll
    for (int j = 0; j < 4; ++j) acc[j] = vzero();
    gemm_rowtiles<4>(acc, &sxb[0][0], 1024, m, wf1t, 1024, nlane, 1024, hf);
#pragma unroll
    for (int j = 0; j < 4; ++j) {
      int n = nlane + j * 16;
      float bias = bf1v[n];
#pragma unroll
      for (int r = 0; r < 8; ++r) shf[r + 8 * hf][n] = f2bf(geluf(acc[j][r] + bias));
    }
  }
  __syncthreads();
  if (tid < 16) {
    float s = bh2[0];
    for (int k = 0; k < 256; ++k) s += shh[tid][k] * wh2[k];
    shalt[tid] = sigm(s);
  }
  __syncthreads();
  // GEMM2 + integrate epilogue: 8 tiles per wave -> 2 passes of NB=4
  for (int pass = 0; pass < 2; ++pass) {
    int nlane = wid * 128 + pass * 64 + (lane & 15);
    v8f acc[4];
#pragma unroll
    for (int j = 0; j < 4; ++j) acc[j] = vzero();
    gemm_rowtiles<4>(acc, &shf[0][0], 2048, m, wf2t, 2048, nlane, 2048, hf);
#pragma unroll
    for (int j = 0; j < 4; ++j) {
      int n = nlane + j * 16;
      float bias = bf2v[n], iwn = iw[n];
#pragma unroll
      for (int r = 0; r < 8; ++r) {
        int mm = r + 8 * hf;
        long long gg = (long long)(t0 + mm) * D_MODEL + n;
        integ[gg] += shalt[mm] * (acc[j][r] + bias) * iwn;
      }
    }
  }
}

// ---------------------------------------------------------------------------
extern "C" void kernel_launch(void* const* d_in, const int* in_sizes, int n_in,
                              void* d_out, int out_size, void* d_ws, size_t ws_size,
                              hipStream_t stream) {
  const float* x    = (const float*)d_in[0];
  const float* w_r1 = (const float*)d_in[1];
  const float* b_r1 = (const float*)d_in[2];
  const float* w_r2 = (const float*)d_in[3];
  const float* b_r2 = (const float*)d_in[4];
  const float* w_h1 = (const float*)d_in[5];
  const float* b_h1 = (const float*)d_in[6];
  const float* w_h2 = (const float*)d_in[7];
  const float* b_h2 = (const float*)d_in[8];
  const float* e_w1 = (const float*)d_in[9];
  const float* e_b1 = (const float*)d_in[10];
  const float* e_w2 = (const float*)d_in[11];
  const float* e_b2 = (const float*)d_in[12];
  const float* w_s1 = (const float*)d_in[13];
  const float* b_s1 = (const float*)d_in[14];
  const float* w_s2 = (const float*)d_in[15];
  const float* b_s2 = (const float*)d_in[16];
  const float* shw  = (const float*)d_in[17];
  const float* w_f1 = (const float*)d_in[18];
  const float* b_f1 = (const float*)d_in[19];
  const float* w_f2 = (const float*)d_in[20];
  const float* b_f2 = (const float*)d_in[21];
  const float* iw   = (const float*)d_in[22];
  const float* mu   = (const float*)d_in[23];
  float* out = (float*)d_out;
  const int T = in_sizes[0] / D_MODEL;   // 16384

  char* ws = (char*)d_ws;
  size_t off = 0;
  auto take = [&](size_t bytes) -> char* {
    off = (off + 255) & ~(size_t)255;
    char* p = ws + off;
    off += bytes;
    return p;
  };
  float* v_buf           = (float*)take((size_t)T * D_MODEL * 4);
  unsigned short* xn_buf = (unsigned short*)take((size_t)T * D_MODEL * 2);
  float* wexp            = (float*)take((size_t)T * NEXP * 4);
  unsigned short* w_r1t = (unsigned short*)take((size_t)256 * 1024 * 2);
  unsigned short* w_h1t = (unsigned short*)take((size_t)256 * 1024 * 2);
  unsigned short* w_s1t = (unsigned short*)take((size_t)64 * 2048 * 2);
  unsigned short* w_s2t = (unsigned short*)take((size_t)3072 * 64 * 2);
  unsigned short* e_w1t = (unsigned short*)take((size_t)8 * 64 * 2048 * 2);
  unsigned short* e_w2t = (unsigned short*)take((size_t)8 * 3072 * 64 * 2);
  unsigned short* w_f1t = (unsigned short*)take((size_t)2048 * 1024 * 2);
  unsigned short* w_f2t = (unsigned short*)take((size_t)1024 * 2048 * 2);

  auto tlaunch = [&](const float* in, unsigned short* o, int B, int R, int C) {
    long long total = (long long)B * R * C;
    int blocks = (int)((total + 255) / 256);
    hipLaunchKernelGGL(k_transpose_bf16, dim3(blocks), dim3(256), 0, stream, in, o, B, R, C);
  };
  tlaunch(w_r1, w_r1t, 1, 1024, 256);
  tlaunch(w_h1, w_h1t, 1, 1024, 256);
  tlaunch(w_s1, w_s1t, 1, 2048, 64);
  tlaunch(w_s2, w_s2t, 1, 64, 3072);
  tlaunch(e_w1, e_w1t, 8, 2048, 64);
  tlaunch(e_w2, e_w2t, 8, 64, 3072);
  tlaunch(w_f1, w_f1t, 1, 1024, 2048);
  tlaunch(w_f2, w_f2t, 1, 2048, 1024);

  const int nblk = T / 16;
  hipLaunchKernelGGL(k_router_init, dim3(nblk), dim3(256), 0, stream,
                     x, w_r1t, b_r1, w_r2, b_r2, out, v_buf, wexp);
  for (int it = 0; it < 2; ++it) {
    hipLaunchKernelGGL(k_experts, dim3(nblk), dim3(256), 0, stream,
                       out, v_buf, xn_buf, w_s1t, b_s1, w_s2t, b_s2,
                       e_w1t, e_b1, e_w2t, e_b2, wexp, shw, mu);
    hipLaunchKernelGGL(k_refine, dim3(nblk), dim3(256), 0, stream,
                       xn_buf, out, w_h1t, b_h1, w_h2, b_h2,
                       w_f1t, b_f1, w_f2t, b_f2, iw);
  }
  (void)n_in; (void)out_size; (void)ws_size;
}
